// CoreHybridBlock_76459007803787
// MI455X (gfx1250) — matrix operand order repaired and checked by reading, not compiled, not run on hardware
//
#include <hip/hip_runtime.h>
#include <hip/hip_bf16.h>

typedef _Float16 half_t;
typedef __attribute__((ext_vector_type(16))) _Float16 v16h;
typedef __attribute__((ext_vector_type(8)))  _Float16 v8h;
typedef __attribute__((ext_vector_type(8)))  float    v8f;

#define B_SZ    8
#define L_SEQ   4096
#define D_MODEL 512
#define D_CONV  256
#define D_MAMBA 256
#define DSTATE  64
#define KCONV   3
#define FFN     2048
#define M_TOK   (B_SZ * L_SEQ)      // 32768
#define NCAT    640                 // 256 xssm | 256 dt | 64 B | 64 C

#define LDSS 40                     // LDS row stride in halfs (16B aligned, 80B)

// ---- CDNA5 async global->LDS support (guarded; falls back to sync copies) --
#if defined(__has_builtin)
#  if __has_builtin(__builtin_amdgcn_global_load_async_to_lds_b128)
#    define HAVE_ASYNC_LDS 1
#  endif
#endif
#ifndef HAVE_ASYNC_LDS
#  define HAVE_ASYNC_LDS 0
#endif

// One B128 beat: 16-byte int vector, matching the builtin's pointee type.
typedef int v4i __attribute__((vector_size(16)));
typedef __attribute__((address_space(1))) v4i g_v4i;   // global
typedef __attribute__((address_space(3))) v4i l_v4i;   // LDS

__device__ __forceinline__ g_v4i* to_gas(const void* p) {
  // AS1 pointer representation == 64-bit VA
  return (g_v4i*)(unsigned long long)p;
}
__device__ __forceinline__ l_v4i* to_las(const void* p) {
  // generic LDS address: low 32 bits are the LDS byte offset (ISA 10.2)
  return (l_v4i*)(unsigned)(unsigned long long)p;
}

__device__ __forceinline__ void async_b128(const half_t* g, half_t* l) {
#if HAVE_ASYNC_LDS
  // inst offset is added to BOTH the global and LDS addresses (ISA 10, async)
  __builtin_amdgcn_global_load_async_to_lds_b128(to_gas(g), to_las(l), 0, 0);
  __builtin_amdgcn_global_load_async_to_lds_b128(to_gas(g), to_las(l), 16, 0);
#else
  *(v8h*)l       = *(const v8h*)g;
  *(v8h*)(l + 8) = *(const v8h*)(g + 8);
#endif
}

__device__ __forceinline__ void wait_async_lds() {
#if HAVE_ASYNC_LDS
#  if defined(__has_builtin) && __has_builtin(__builtin_amdgcn_s_wait_asynccnt)
  __builtin_amdgcn_s_wait_asynccnt(0);
#  else
  asm volatile("s_wait_asynccnt 0x0" ::: "memory");
#  endif
#endif
}

// ---------------------------------------------------------------------------
// Weight convert + transpose: src f32 (K x N) row-major -> dst f16 (N x K)
// ---------------------------------------------------------------------------
__global__ __launch_bounds__(256) void k_cvt_t(const float* __restrict__ src,
                                               half_t* __restrict__ dst,
                                               int K, int N) {
  int idx = blockIdx.x * 256 + threadIdx.x;
  if (idx >= K * N) return;
  int k = idx / N, n = idx % N;
  dst[(size_t)n * K + k] = (half_t)src[idx];
}

// ---------------------------------------------------------------------------
// RMSNorm: x (M_TOK x 512) f32 -> out f16, one block per row
// ---------------------------------------------------------------------------
__global__ __launch_bounds__(256) void k_rmsnorm(const float* __restrict__ x,
                                                 const float* __restrict__ w,
                                                 half_t* __restrict__ out) {
  int row = blockIdx.x;
  int tid = threadIdx.x;
  const float* xr = x + (size_t)row * D_MODEL;
  float v0 = xr[tid], v1 = xr[tid + 256];
  __shared__ float red[256];
  red[tid] = v0 * v0 + v1 * v1;
  __syncthreads();
  for (int st = 128; st > 0; st >>= 1) {
    if (tid < st) red[tid] += red[tid + st];
    __syncthreads();
  }
  float r = rsqrtf(red[0] * (1.0f / D_MODEL) + 1e-6f);
  half_t* o = out + (size_t)row * D_MODEL;
  o[tid]       = (half_t)(v0 * r * w[tid]);
  o[tid + 256] = (half_t)(v1 * r * w[tid + 256]);
}

// ---------------------------------------------------------------------------
// WMMA GEMM: C[M x N] = A[M x K] * Bt[N x K]^T (f16 in, f32 accumulate).
// Block tile 128x128, 8 waves, wave tile 32x64 (2x4 WMMA tiles).
// Double-buffered LDS; tiles staged with GLOBAL_LOAD_ASYNC_TO_LDS_B128 and
// pipelined one K-step ahead (wait asynccnt + one barrier per step).
// FLAGS: bit0 -> f16 output, bit1 -> add addsrc (f32) before store.
// Requires: M%128==0, N%128==0, K%32==0.
// ---------------------------------------------------------------------------
template <int FLAGS>
__global__ __launch_bounds__(256) void k_gemm(const half_t* __restrict__ A,
                                              const half_t* __restrict__ Bt,
                                              float* __restrict__ Cf,
                                              half_t* __restrict__ Ch,
                                              const float* __restrict__ addsrc,
                                              int M, int N, int K,
                                              int ldc, int col_off) {
  __shared__ half_t As[2][128 * LDSS];
  __shared__ half_t Bs[2][128 * LDSS];
  const int tid  = threadIdx.x;
  const int lane = tid & 31;
  const int wave = tid >> 5;
  const int wm = wave >> 1;      // 0..3
  const int wn = wave & 1;       // 0..1
  const int llo = lane & 15;
  const int lhi = lane >> 4;
  const int m0 = blockIdx.y * 128;
  const int n0 = blockIdx.x * 128;
  const int tr = tid >> 1;           // row 0..127 for cooperative loads
  const int tk = (tid & 1) * 16;     // half-offset 0/16

  v8f acc[2][4];
  const v8f zer = {0.f, 0.f, 0.f, 0.f, 0.f, 0.f, 0.f, 0.f};
#pragma unroll
  for (int i = 0; i < 2; ++i)
#pragma unroll
    for (int j = 0; j < 4; ++j) acc[i][j] = zer;

  auto load_tile = [&](int buf, int k0) {
    const half_t* ag = A  + (size_t)(m0 + tr) * K + k0 + tk;
    const half_t* bg = Bt + (size_t)(n0 + tr) * K + k0 + tk;
    async_b128(ag, &As[buf][tr * LDSS + tk]);
    async_b128(bg, &Bs[buf][tr * LDSS + tk]);
  };

  const int nk = K >> 5;
  load_tile(0, 0);
  for (int i = 0; i < nk; ++i) {
    const int p = i & 1;
    wait_async_lds();        // this wave's tile-i transfers landed in LDS
    __syncthreads();         // all waves' transfers landed; prior readers done
    if (i + 1 < nk) load_tile(1 - p, (i + 1) << 5);

    v16h afrag[2], bfrag[4];
#pragma unroll
    for (int tm = 0; tm < 2; ++tm) {
      // A 16x32 f16 frag: lane -> row M=llo; halves j -> K = lhi*16 + j
      const half_t* ap = &As[p][(wm * 32 + tm * 16 + llo) * LDSS];
      v8h lo = *(const v8h*)(ap + lhi * 16);
      v8h hi = *(const v8h*)(ap + lhi * 16 + 8);
#pragma unroll
      for (int q = 0; q < 8; ++q) { afrag[tm][q] = lo[q]; afrag[tm][8 + q] = hi[q]; }
    }
#pragma unroll
    for (int tn = 0; tn < 4; ++tn) {
      // B 32x16 f16 frag: lane -> col N=llo; halves j -> K = lhi*16 + j
      const half_t* bp = &Bs[p][(wn * 64 + tn * 16 + llo) * LDSS];
      v8h lo = *(const v8h*)(bp + lhi * 16);
      v8h hi = *(const v8h*)(bp + lhi * 16 + 8);
#pragma unroll
      for (int q = 0; q < 8; ++q) { bfrag[tn][q] = lo[q]; bfrag[tn][8 + q] = hi[q]; }
    }
#pragma unroll
    for (int tm = 0; tm < 2; ++tm)
#pragma unroll
      for (int tn = 0; tn < 4; ++tn)
        acc[tm][tn] = __builtin_amdgcn_wmma_f32_16x16x32_f16(
            false, afrag[tm], false, bfrag[tn], (short)0, acc[tm][tn], false, false);
  }

  // C/D frag: VGPR v, lanes 0-15 -> M=v, lanes 16-31 -> M=8+v; N=llo
#pragma unroll
  for (int tm = 0; tm < 2; ++tm)
#pragma unroll
    for (int tn = 0; tn < 4; ++tn) {
      int rbase = m0 + wm * 32 + tm * 16 + lhi * 8;
      int col   = n0 + wn * 64 + tn * 16 + llo;
#pragma unroll
      for (int v = 0; v < 8; ++v) {
        size_t idx = (size_t)(rbase + v) * ldc + col_off + col;
        float val = acc[tm][tn][v];
        if constexpr ((FLAGS & 2) != 0) val += addsrc[idx];
        if constexpr ((FLAGS & 1) != 0) Ch[idx] = (half_t)val;
        else                            Cf[idx] = val;
      }
    }
}

// ---------------------------------------------------------------------------
// Fused dual WMMA GEMM for SwiGLU: act = silu(A*W1t^T) * (A*W3t^T), f16 out
// ---------------------------------------------------------------------------
__global__ __launch_bounds__(256) void k_gemm_gated(const half_t* __restrict__ A,
                                                    const half_t* __restrict__ B1t,
                                                    const half_t* __restrict__ B3t,
                                                    half_t* __restrict__ act,
                                                    int M, int N, int K) {
  __shared__ half_t As [2][128 * LDSS];
  __shared__ half_t B1s[2][128 * LDSS];
  __shared__ half_t B3s[2][128 * LDSS];
  const int tid  = threadIdx.x;
  const int lane = tid & 31;
  const int wave = tid >> 5;
  const int wm = wave >> 1, wn = wave & 1;
  const int llo = lane & 15, lhi = lane >> 4;
  const int m0 = blockIdx.y * 128, n0 = blockIdx.x * 128;
  const int tr = tid >> 1, tk = (tid & 1) * 16;

  v8f acc1[2][4], acc3[2][4];
  const v8f zer = {0.f, 0.f, 0.f, 0.f, 0.f, 0.f, 0.f, 0.f};
#pragma unroll
  for (int i = 0; i < 2; ++i)
#pragma unroll
    for (int j = 0; j < 4; ++j) { acc1[i][j] = zer; acc3[i][j] = zer; }

  auto load_tile = [&](int buf, int k0) {
    const half_t* ag = A   + (size_t)(m0 + tr) * K + k0 + tk;
    const half_t* g1 = B1t + (size_t)(n0 + tr) * K + k0 + tk;
    const half_t* g3 = B3t + (size_t)(n0 + tr) * K + k0 + tk;
    async_b128(ag, &As [buf][tr * LDSS + tk]);
    async_b128(g1, &B1s[buf][tr * LDSS + tk]);
    async_b128(g3, &B3s[buf][tr * LDSS + tk]);
  };

  const int nk = K >> 5;
  load_tile(0, 0);
  for (int i = 0; i < nk; ++i) {
    const int p = i & 1;
    wait_async_lds();
    __syncthreads();
    if (i + 1 < nk) load_tile(1 - p, (i + 1) << 5);

    v16h afrag[2], b1f[4], b3f[4];
#pragma unroll
    for (int tm = 0; tm < 2; ++tm) {
      const half_t* ap = &As[p][(wm * 32 + tm * 16 + llo) * LDSS];
      v8h lo = *(const v8h*)(ap + lhi * 16);
      v8h hi = *(const v8h*)(ap + lhi * 16 + 8);
#pragma unroll
      for (int q = 0; q < 8; ++q) { afrag[tm][q] = lo[q]; afrag[tm][8 + q] = hi[q]; }
    }
#pragma unroll
    for (int tn = 0; tn < 4; ++tn) {
      const half_t* p1 = &B1s[p][(wn * 64 + tn * 16 + llo) * LDSS];
      const half_t* p3 = &B3s[p][(wn * 64 + tn * 16 + llo) * LDSS];
      v8h l1 = *(const v8h*)(p1 + lhi * 16);
      v8h h1 = *(const v8h*)(p1 + lhi * 16 + 8);
      v8h l3 = *(const v8h*)(p3 + lhi * 16);
      v8h h3 = *(const v8h*)(p3 + lhi * 16 + 8);
#pragma unroll
      for (int q = 0; q < 8; ++q) {
        b1f[tn][q] = l1[q]; b1f[tn][8 + q] = h1[q];
        b3f[tn][q] = l3[q]; b3f[tn][8 + q] = h3[q];
      }
    }
#pragma unroll
    for (int tm = 0; tm < 2; ++tm)
#pragma unroll
      for (int tn = 0; tn < 4; ++tn) {
        acc1[tm][tn] = __builtin_amdgcn_wmma_f32_16x16x32_f16(
            false, afrag[tm], false, b1f[tn], (short)0, acc1[tm][tn], false, false);
        acc3[tm][tn] = __builtin_amdgcn_wmma_f32_16x16x32_f16(
            false, afrag[tm], false, b3f[tn], (short)0, acc3[tm][tn], false, false);
      }
  }

#pragma unroll
  for (int tm = 0; tm < 2; ++tm)
#pragma unroll
    for (int tn = 0; tn < 4; ++tn) {
      int rbase = m0 + wm * 32 + tm * 16 + lhi * 8;
      int col   = n0 + wn * 64 + tn * 16 + llo;
#pragma unroll
      for (int v = 0; v < 8; ++v) {
        float h1 = acc1[tm][tn][v];
        float h3 = acc3[tm][tn][v];
        float s  = h1 / (1.0f + expf(-h1));
        act[(size_t)(rbase + v) * N + col] = (half_t)(s * h3);
      }
    }
}

// ---------------------------------------------------------------------------
// SSM prep: per token, from packed proj row [xssm(256)|dt(256)|B(64)|C(64)]
// -> decay, inp (256), Cn (64)
// ---------------------------------------------------------------------------
__global__ __launch_bounds__(256) void k_ssm_prep(const float* __restrict__ cat,
                                                  const float* __restrict__ dt_b,
                                                  const float* __restrict__ A_log,
                                                  float* __restrict__ decay,
                                                  float* __restrict__ inp,
                                                  float* __restrict__ Cn) {
  int t = blockIdx.x;
  int c = threadIdx.x;
  const float* rowp = cat + (size_t)t * NCAT;
  __shared__ float sB[64], sC[64], sred[64];
  __shared__ float nB, nC;
  if (c < 64) { sB[c] = rowp[512 + c]; sC[c] = rowp[576 + c]; }
  __syncthreads();
  if (c < 64) sred[c] = sB[c] * sB[c];
  __syncthreads();
  for (int st = 32; st > 0; st >>= 1) {
    if (c < st) sred[c] += sred[c + st];
    __syncthreads();
  }
  if (c == 0) nB = fmaxf(sqrtf(sred[0]), 1.0f);
  __syncthreads();
  if (c < 64) sred[c] = sC[c] * sC[c];
  __syncthreads();
  for (int st = 32; st > 0; st >>= 1) {
    if (c < st) sred[c] += sred[c + st];
    __syncthreads();
  }
  if (c == 0) nC = fmaxf(sqrtf(sred[0]), 1.0f);
  __syncthreads();

  float dtr = rowp[256 + c] + dt_b[c];
  dtr = fminf(fmaxf(dtr, -10.0f), 5.0f);
  float dt = log1pf(expf(dtr));              // softplus
  dt = fminf(fmaxf(dt, 1e-4f), 0.1f);
  float Ac  = -expf(A_log[c]);
  float bn  = sB[c & 63] / nB;
  float xs  = rowp[c];
  decay[(size_t)t * D_MAMBA + c] = expf(dt * Ac);
  inp  [(size_t)t * D_MAMBA + c] = dt * bn * xs;
  if (c < 64) Cn[(size_t)t * 64 + c] = sC[c] / nC;
}

// ---------------------------------------------------------------------------
// Sequential SSM scan over L per (batch, channel); y -> f16.
// Latency-bound serial recurrence -> prefetch a few steps ahead
// (global_prefetch_b8) so each step's operands are already in cache.
// ---------------------------------------------------------------------------
__global__ __launch_bounds__(256) void k_scan(const float* __restrict__ decay,
                                              const float* __restrict__ inp,
                                              const float* __restrict__ Cn,
                                              const float* __restrict__ cat,
                                              const float* __restrict__ Dv,
                                              half_t* __restrict__ y) {
  int b = blockIdx.x;      // 0..7
  int c = threadIdx.x;     // 0..255
  float h = 0.0f;
  float dc = Dv[c];
  for (int l = 0; l < L_SEQ; ++l) {
    size_t t = (size_t)b * L_SEQ + l;
    if (l + 8 < L_SEQ) {
      size_t tp = t + 8;
      __builtin_prefetch(&decay[tp * D_MAMBA + c], 0, 0);
      __builtin_prefetch(&inp  [tp * D_MAMBA + c], 0, 0);
      __builtin_prefetch(&cat  [tp * NCAT    + c], 0, 0);
    }
    h = decay[t * D_MAMBA + c] * h + inp[t * D_MAMBA + c];
    float yv = h * Cn[t * 64 + (c & 63)] + dc * cat[t * NCAT + c];
    y[t * D_MAMBA + c] = (half_t)yv;
  }
}

// ---------------------------------------------------------------------------
// Depthwise causal conv (K=3) + SiLU gate; writes cols [0,256) of mixcat f16
// ---------------------------------------------------------------------------
__global__ __launch_bounds__(256) void k_conv_gate(const float* __restrict__ uv,
                                                   const float* __restrict__ dww,
                                                   const float* __restrict__ dwb,
                                                   half_t* __restrict__ mixcat) {
  int t = blockIdx.x;          // token
  int c = threadIdx.x;         // channel 0..255
  int b = t >> 12;             // /4096
  int l = t & (L_SEQ - 1);
  float conv = dwb[c];
#pragma unroll
  for (int k = 0; k < KCONV; ++k) {
    int lk = l - (KCONV - 1) + k;
    if (lk >= 0)
      conv += uv[((size_t)b * L_SEQ + lk) * D_MODEL + c] * dww[c * KCONV + k];
  }
  float g = uv[(size_t)t * D_MODEL + D_CONV + c];
  float sg = g / (1.0f + expf(-g));
  mixcat[(size_t)t * D_MODEL + c] = (half_t)(conv * sg);
}

// ---------------------------------------------------------------------------
// velocity = sigmoid(log_beta)*velocity + mixer; xmid = x + velocity;
// write velocity to out_v, xmid to ws; normed = rmsnorm(xmid)*ffn_w -> f16
// ---------------------------------------------------------------------------
__global__ __launch_bounds__(256) void k_resid_norm(const float* __restrict__ x,
                                                    const float* __restrict__ vel,
                                                    const float* __restrict__ mixer,
                                                    const float* __restrict__ log_beta,
                                                    const float* __restrict__ ffn_w,
                                                    float* __restrict__ out_v,
                                                    float* __restrict__ xmid,
                                                    half_t* __restrict__ normed) {
  int row = blockIdx.x;
  int tid = threadIdx.x;
  float beta = 1.0f / (1.0f + expf(-log_beta[0]));
  size_t base = (size_t)row * D_MODEL;
  float m0 = mixer[base + tid],       m1 = mixer[base + tid + 256];
  float v0 = beta * vel[base + tid] + m0;
  float v1 = beta * vel[base + tid + 256] + m1;
  out_v[base + tid] = v0;  out_v[base + tid + 256] = v1;
  float x0 = x[base + tid] + v0;
  float x1 = x[base + tid + 256] + v1;
  xmid[base + tid] = x0;  xmid[base + tid + 256] = x1;
  __shared__ float red[256];
  red[tid] = x0 * x0 + x1 * x1;
  __syncthreads();
  for (int st = 128; st > 0; st >>= 1) {
    if (tid < st) red[tid] += red[tid + st];
    __syncthreads();
  }
  float r = rsqrtf(red[0] * (1.0f / D_MODEL) + 1e-6f);
  normed[base + tid]       = (half_t)(x0 * r * ffn_w[tid]);
  normed[base + tid + 256] = (half_t)(x1 * r * ffn_w[tid + 256]);
}

// ---------------------------------------------------------------------------
extern "C" void kernel_launch(void* const* d_in, const int* in_sizes, int n_in,
                              void* d_out, int out_size, void* d_ws, size_t ws_size,
                              hipStream_t stream) {
  const float* x         = (const float*)d_in[0];
  const float* velocity  = (const float*)d_in[1];
  const float* pre_w     = (const float*)d_in[2];
  const float* conv_in_w = (const float*)d_in[3];
  const float* dw_w      = (const float*)d_in[4];
  const float* dw_b      = (const float*)d_in[5];
  const float* x_proj_w  = (const float*)d_in[6];
  const float* dt_w      = (const float*)d_in[7];
  const float* dt_b      = (const float*)d_in[8];
  const float* B_w       = (const float*)d_in[9];
  const float* C_w       = (const float*)d_in[10];
  const float* A_log     = (const float*)d_in[11];
  const float* Dv        = (const float*)d_in[12];
  const float* ssm_out_w = (const float*)d_in[13];
  const float* out_projw = (const float*)d_in[14];
  const float* log_beta  = (const float*)d_in[15];
  const float* ffn_w     = (const float*)d_in[16];
  const float* w1        = (const float*)d_in[17];
  const float* w3        = (const float*)d_in[18];
  const float* w2        = (const float*)d_in[19];
  (void)in_sizes; (void)n_in; (void)out_size; (void)ws_size;

  char* ws = (char*)d_ws;
  size_t off = 0;
  auto alloc = [&](size_t bytes) { char* p = ws + off; off += (bytes + 255) & ~(size_t)255; return p; };

  half_t* convinT = (half_t*)alloc((size_t)D_MODEL * D_MODEL * 2);   // 512x512
  half_t* catT    = (half_t*)alloc((size_t)NCAT    * D_MODEL * 2);   // 640x512
  half_t* ssmoT   = (half_t*)alloc((size_t)D_MAMBA * D_MAMBA * 2);   // 256x256
  half_t* outpT   = (half_t*)alloc((size_t)D_MODEL * D_MODEL * 2);   // 512x512
  half_t* w1T     = (half_t*)alloc((size_t)FFN     * D_MODEL * 2);   // 2048x512
  half_t* w3T     = (half_t*)alloc((size_t)FFN     * D_MODEL * 2);
  half_t* w2T     = (half_t*)alloc((size_t)D_MODEL * FFN     * 2);   // 512x2048
  half_t* xn_h    = (half_t*)alloc((size_t)M_TOK * D_MODEL * 2);
  float*  uv      = (float*) alloc((size_t)M_TOK * D_MODEL * 4);     // reused as mixer
  float*  cat_o   = (float*) alloc((size_t)M_TOK * NCAT    * 4);
  float*  decay   = (float*) alloc((size_t)M_TOK * D_MAMBA * 4);
  float*  inp     = (float*) alloc((size_t)M_TOK * D_MAMBA * 4);
  float*  Cn      = (float*) alloc((size_t)M_TOK * DSTATE  * 4);
  half_t* y_h     = (half_t*)alloc((size_t)M_TOK * D_MAMBA * 2);
  half_t* mixcat  = (half_t*)alloc((size_t)M_TOK * D_MODEL * 2);
  float*  xmid    = (float*) alloc((size_t)M_TOK * D_MODEL * 4);
  half_t* norm_h  = (half_t*)alloc((size_t)M_TOK * D_MODEL * 2);
  half_t* act_h   = (half_t*)alloc((size_t)M_TOK * FFN     * 2);
  float*  mixer   = uv;   // uv dead after conv_gate; reuse for mixer

  float* out_x = (float*)d_out;
  float* out_v = out_x + (size_t)M_TOK * D_MODEL;

  auto cvt = [&](const float* src, half_t* dst, int K, int N) {
    int n = K * N;
    k_cvt_t<<<(n + 255) / 256, 256, 0, stream>>>(src, dst, K, N);
  };
  // Weights -> f16, transposed (N x K). Pack x_proj|dt|B|C into catT rows.
  cvt(conv_in_w, convinT, D_MODEL, D_MODEL);
  cvt(x_proj_w,  catT,                         D_MODEL, D_MAMBA);
  cvt(dt_w,      catT + (size_t)256 * D_MODEL, D_MODEL, D_MAMBA);
  cvt(B_w,       catT + (size_t)512 * D_MODEL, D_MODEL, DSTATE);
  cvt(C_w,       catT + (size_t)576 * D_MODEL, D_MODEL, DSTATE);
  cvt(ssm_out_w, ssmoT, D_MAMBA, D_MAMBA);
  cvt(out_projw, outpT, D_MODEL, D_MODEL);
  cvt(w1, w1T, D_MODEL, FFN);
  cvt(w3, w3T, D_MODEL, FFN);
  cvt(w2, w2T, FFN, D_MODEL);

  // 1) xn = rmsnorm(x) -> f16
  k_rmsnorm<<<M_TOK, 256, 0, stream>>>(x, pre_w, xn_h);

  // 2) uv = xn @ conv_in  (N=512)
  k_gemm<0><<<dim3(D_MODEL / 128, M_TOK / 128), 256, 0, stream>>>(
      xn_h, convinT, uv, nullptr, nullptr, M_TOK, D_MODEL, D_MODEL, D_MODEL, 0);

  // 3) packed SSM projections (N=640): [xssm|dt|B|C]
  k_gemm<0><<<dim3(NCAT / 128, M_TOK / 128), 256, 0, stream>>>(
      xn_h, catT, cat_o, nullptr, nullptr, M_TOK, NCAT, D_MODEL, NCAT, 0);

  // 4) SSM prep + scan
  k_ssm_prep<<<M_TOK, 256, 0, stream>>>(cat_o, dt_b, A_log, decay, inp, Cn);
  k_scan<<<B_SZ, 256, 0, stream>>>(decay, inp, Cn, cat_o, Dv, y_h);

  // 5) conv+gate -> mixcat[:, 0:256)
  k_conv_gate<<<M_TOK, 256, 0, stream>>>(uv, dw_w, dw_b, mixcat);

  // 6) ssm_out = y @ ssm_out_w -> mixcat[:, 256:512) as f16
  k_gemm<1><<<dim3(D_MAMBA / 128, M_TOK / 128), 256, 0, stream>>>(
      y_h, ssmoT, nullptr, mixcat, nullptr, M_TOK, D_MAMBA, D_MAMBA, D_MODEL, 256);

  // 7) mixer = mixcat @ out_proj (reuses uv buffer)
  k_gemm<0><<<dim3(D_MODEL / 128, M_TOK / 128), 256, 0, stream>>>(
      mixcat, outpT, mixer, nullptr, nullptr, M_TOK, D_MODEL, D_MODEL, D_MODEL, 0);

  // 8) velocity/residual update + ffn rmsnorm
  k_resid_norm<<<M_TOK, 256, 0, stream>>>(x, velocity, mixer, log_beta, ffn_w,
                                          out_v, xmid, norm_h);

  // 9) act = silu(normed@w1) * (normed@w3), fused dual-GEMM
  k_gemm_gated<<<dim3(FFN / 128, M_TOK / 128), 256, 0, stream>>>(
      norm_h, w1T, w3T, act_h, M_TOK, FFN, D_MODEL);

  // 10) x_out = xmid + act @ w2  (fused residual add into GEMM epilogue)
  k_gemm<2><<<dim3(D_MODEL / 128, M_TOK / 128), 256, 0, stream>>>(
      act_h, w2T, out_x, nullptr, xmid, M_TOK, D_MODEL, FFN, D_MODEL, 0);
}
